// DualPrompt_32804960207309
// MI455X (gfx1250) — matrix operand order repaired and checked by reading, not compiled, not run on hardware
//
#include <hip/hip_runtime.h>
#include <hip/hip_bf16.h>

// ---------------- types ----------------
typedef _Float16 half_t;
typedef __attribute__((ext_vector_type(16))) _Float16 v16h;
typedef __attribute__((ext_vector_type(8)))  float    v8f;
typedef int dp_v4i __attribute__((vector_size(16)));   // matches builtin param type

#define B_   64
#define L_   12
#define H_   12
#define D_   768
#define FF_  3072
#define NT_  197
#define NP_  196
#define NC_  100
#define HD_  64
#define M1_  (B_ * NT_)          // 12608 real rows (B*tokens)
#define MPAD_ 12672              // 99 * 128 (padded for unguarded 128-tiles)
#define MP_  (B_ * NP_)          // 12544 = 98*128 rows (B*patches)

// ---- optional gfx1250 async global->LDS path (compile-safe gate) ----
#ifndef DP_TRY_ASYNC
#define DP_TRY_ASYNC 1
#endif
#if DP_TRY_ASYNC && defined(__has_builtin)
#if __has_builtin(__builtin_amdgcn_global_load_async_to_lds_b128) && \
    __has_builtin(__builtin_amdgcn_s_wait_asynccnt)
#define DP_HAS_ASYNC 1
#endif
#endif
#ifndef DP_HAS_ASYNC
#define DP_HAS_ASYNC 0
#endif

// ============================================================
// Generic f32 -> f16 conversion (+ zero-padded variant)
// ============================================================
__global__ void dp_cvt_f32_to_f16(const float* __restrict__ s, half_t* __restrict__ d, long long n) {
    long long i = (long long)blockIdx.x * blockDim.x + threadIdx.x;
    if (i < n) d[i] = (half_t)s[i];
}
__global__ void dp_cvt_pad(const float* __restrict__ s, half_t* __restrict__ d,
                           long long nsrc, long long ntot) {
    long long i = (long long)blockIdx.x * blockDim.x + threadIdx.x;
    if (i < ntot) d[i] = (i < nsrc) ? (half_t)s[i] : (half_t)0.f;
}
__global__ void dp_pad_f32(const float* __restrict__ s, float* __restrict__ d,
                           int nsrc, int ntot) {
    int i = blockIdx.x * blockDim.x + threadIdx.x;
    if (i < ntot) d[i] = (i < nsrc) ? s[i] : 0.f;
}

// ============================================================
// im2col for 16x16 stride-16 patches: [B,3,224,224] -> [B*196, 768] f16
// ============================================================
__global__ void dp_im2col(const float* __restrict__ in, half_t* __restrict__ out) {
    long long i = (long long)blockIdx.x * blockDim.x + threadIdx.x;
    const long long total = (long long)B_ * NP_ * 768;
    if (i >= total) return;
    int f = (int)(i % 768);
    long long r = i / 768;
    int p = (int)(r % NP_);
    int b = (int)(r / NP_);
    int c  = f >> 8;
    int py = (f >> 4) & 15;
    int px = f & 15;
    int gy = p / 14, gx = p % 14;
    long long src = (((long long)b * 3 + c) * 224 + (gy * 16 + py)) * 224 + (gx * 16 + px);
    out[i] = (half_t)in[src];
}

// ============================================================
// build tokens: tok_pe[b,t,d] = (t==0 ? cls[d] : patches16[b,t-1,d]) + pos[t,d]
// ============================================================
__global__ void dp_build_tokens(const half_t* __restrict__ patches,
                                const float* __restrict__ cls,
                                const float* __restrict__ pos,
                                float* __restrict__ tok_pe) {
    long long i = (long long)blockIdx.x * blockDim.x + threadIdx.x;
    const long long total = (long long)M1_ * D_;
    if (i >= total) return;
    int d = (int)(i % D_);
    long long r = i / D_;
    int t = (int)(r % NT_);
    int b = (int)(r / NT_);
    float v;
    if (t == 0) v = cls[d];
    else        v = (float)patches[((long long)b * NP_ + (t - 1)) * D_ + d];
    tok_pe[i] = v + pos[(long long)t * D_ + d];
}

// ============================================================
// LayerNorm over D=768. One block (256 thr) per row.
// ============================================================
__global__ void dp_layernorm(const float* __restrict__ x, long long rowStride,
                             const float* __restrict__ g, const float* __restrict__ bta,
                             half_t* __restrict__ oh, long long ohStride,
                             float* __restrict__ of, long long ofStride) {
    const int row = blockIdx.x;
    const int tid = threadIdx.x;
    const float* xr = x + (long long)row * rowStride;
    __shared__ float s1[256], s2[256];
    float a = 0.f, b2 = 0.f;
    for (int i = tid; i < D_; i += 256) { float v = xr[i]; a += v; b2 += v * v; }
    s1[tid] = a; s2[tid] = b2; __syncthreads();
    for (int off = 128; off; off >>= 1) {
        if (tid < off) { s1[tid] += s1[tid + off]; s2[tid] += s2[tid + off]; }
        __syncthreads();
    }
    float mean = s1[0] * (1.0f / D_);
    float var  = s2[0] * (1.0f / D_) - mean * mean;
    float rstd = rsqrtf(var + 1e-6f);
    for (int i = tid; i < D_; i += 256) {
        float y = (xr[i] - mean) * rstd * g[i] + bta[i];
        if (oh) oh[(long long)row * ohStride + i] = (half_t)y;
        if (of) of[(long long)row * ofStride + i] = y;
    }
}

// ============================================================
// WMMA GEMM: C[M,N] = A[M,K] * B[N,K]^T + bias (+C) (+gelu)
// Fully unguarded; 128x128 tile, 8 waves, each wave 32x64 (2x4 WMMA),
// double-buffered LDS staging (async global->LDS when available).
// ============================================================
__device__ __forceinline__ float dp_gelu(float x) {
    return 0.5f * x * (1.0f + erff(x * 0.70710678118654752f));
}

#if DP_HAS_ASYNC
__device__ __forceinline__ void dp_async_b128(const half_t* g, half_t* l) {
    __builtin_amdgcn_global_load_async_to_lds_b128(
        (__attribute__((address_space(1))) dp_v4i*)(g),
        (__attribute__((address_space(3))) dp_v4i*)(l), 0, 0);
}
#endif

__global__ __launch_bounds__(256)
void dp_gemm_wmma(const half_t* __restrict__ A, int lda,
                  const half_t* __restrict__ Bw, int ldb,
                  const float* __restrict__ bias,
                  float* __restrict__ Cf, half_t* __restrict__ Ch, int ldc,
                  int K, int addC, int act) {
    __shared__ half_t As[2][128][32];
    __shared__ half_t Bs[2][128][32];
    const int tid  = threadIdx.x;
    const int lane = tid & 31;
    const int wave = tid >> 5;
    const long long m0 = (long long)blockIdx.y * 128;
    const long long n0 = (long long)blockIdx.x * 128;
    const int wm = (wave >> 1) * 32;     // wave row offset in tile (0..96)
    const int wn = (wave & 1) * 64;      // wave col offset in tile (0/64)
    const int srow = tid >> 1;           // staging row 0..127
    const int scol = (tid & 1) * 16;     // staging col half 0/16
    const int lm = lane & 15;
    const int lh = lane >> 4;

    const half_t* Ap = A  + (m0 + srow) * lda + scol;
    const half_t* Bp = Bw + (n0 + srow) * ldb + scol;

    v8f c[2][4] = {};

#if DP_HAS_ASYNC
#define DP_STAGE(buf, kk) do {                                   \
        dp_async_b128(Ap + (kk),     &As[buf][srow][scol]);      \
        dp_async_b128(Ap + (kk) + 8, &As[buf][srow][scol + 8]);  \
        dp_async_b128(Bp + (kk),     &Bs[buf][srow][scol]);      \
        dp_async_b128(Bp + (kk) + 8, &Bs[buf][srow][scol + 8]);  \
    } while (0)
#define DP_WAIT() __builtin_amdgcn_s_wait_asynccnt(0)
#else
#define DP_STAGE(buf, kk) do {                                   \
        float4 a0 = *(const float4*)(Ap + (kk));                 \
        float4 a1 = *(const float4*)(Ap + (kk) + 8);             \
        float4 b0 = *(const float4*)(Bp + (kk));                 \
        float4 b1 = *(const float4*)(Bp + (kk) + 8);             \
        *(float4*)&As[buf][srow][scol]     = a0;                 \
        *(float4*)&As[buf][srow][scol + 8] = a1;                 \
        *(float4*)&Bs[buf][srow][scol]     = b0;                 \
        *(float4*)&Bs[buf][srow][scol + 8] = b1;                 \
    } while (0)
#define DP_WAIT()
#endif

    DP_STAGE(0, 0);
    int cur = 0;
    for (int k0 = 0; k0 < K; k0 += 32) {
        DP_WAIT();
        __syncthreads();                  // tile[cur] visible to all waves;
                                          // all waves done reading tile[cur^1]
        if (k0 + 32 < K) DP_STAGE(cur ^ 1, k0 + 32);

        // ---- fragments per ISA 7.12.2 layouts ----
        v16h af[2], bf[4];
        #pragma unroll
        for (int mi = 0; mi < 2; mi++) {
            const int row = wm + mi * 16 + lm;
            v16h a;
            #pragma unroll
            for (int r = 0; r < 8; r++) {
                int kp = (r < 4) ? (lh * 8 + 2 * r) : (16 + lh * 8 + 2 * (r - 4));
                a[2 * r]     = As[cur][row][kp];
                a[2 * r + 1] = As[cur][row][kp + 1];
            }
            af[mi] = a;
        }
        #pragma unroll
        for (int ni = 0; ni < 4; ni++) {
            const int rowb = wn + ni * 16 + lm;
            v16h b;
            #pragma unroll
            for (int j = 0; j < 16; j++) b[j] = Bs[cur][rowb][(j < 8) ? (lh * 8 + j) : (16 + lh * 8 + (j - 8))];
            bf[ni] = b;
        }
        #pragma unroll
        for (int mi = 0; mi < 2; mi++)
            #pragma unroll
            for (int ni = 0; ni < 4; ni++)
                c[mi][ni] = __builtin_amdgcn_wmma_f32_16x16x32_f16(
                    false, af[mi], false, bf[ni], (short)0, c[mi][ni], false, false);
        cur ^= 1;
    }
#undef DP_STAGE
#undef DP_WAIT

    // ---- store: D element r -> row = (lane/16)*8 + r, col = lane%16 ----
    #pragma unroll
    for (int mi = 0; mi < 2; mi++) {
        #pragma unroll
        for (int ni = 0; ni < 4; ni++) {
            const long long row0 = m0 + wm + mi * 16 + lh * 8;
            const long long col  = n0 + wn + ni * 16 + lm;
            const float bv = bias[col];
            #pragma unroll
            for (int r = 0; r < 8; r++) {
                float v = c[mi][ni][r] + bv;
                if (addC) v += Cf[(row0 + r) * ldc + col];
                if (act == 1) v = dp_gelu(v);
                if (Ch) Ch[(row0 + r) * ldc + col] = (half_t)v;
                else    Cf[(row0 + r) * ldc + col] = v;
            }
        }
    }
}

// ============================================================
// Attention: one thread per (b, head, query-token), online softmax.
// ============================================================
__device__ __forceinline__ void dp_attn_accum(const half_t* kb, const half_t* vb,
                                              int kRow, int vRow, int n,
                                              const float* qv, float& m, float& l,
                                              float* acc) {
    for (int j = 0; j < n; j++) {
        const half_t* kp = kb + (long long)j * kRow;
        float s = 0.f;
        #pragma unroll
        for (int d = 0; d < HD_; d++) s += qv[d] * (float)kp[d];
        s *= 0.125f;
        float mn   = fmaxf(m, s);
        float corr = __expf(m - mn);
        float p    = __expf(s - mn);
        l = l * corr + p;
        const half_t* vp = vb + (long long)j * vRow;
        #pragma unroll
        for (int d = 0; d < HD_; d++) acc[d] = acc[d] * corr + p * (float)vp[d];
        m = mn;
    }
}

__global__ __launch_bounds__(256)
void dp_attention(const half_t* __restrict__ q,
                  const half_t* __restrict__ k1, const half_t* __restrict__ v1,
                  const half_t* __restrict__ k2, const half_t* __restrict__ v2,
                  half_t* __restrict__ out,
                  int n1, int n2,
                  int qRow, long long qBatch,
                  int kRow, long long kBatch,
                  int k2Row, long long k2Batch,
                  int oRow, long long oBatch) {
    int gid = blockIdx.x * blockDim.x + threadIdx.x;
    const int total = B_ * H_ * NT_;
    if (gid >= total) return;
    int t  = gid % NT_;
    int bh = gid / NT_;
    int h  = bh % H_;
    int b  = bh / H_;

    const half_t* qp = q + (long long)b * qBatch + (long long)t * qRow + h * HD_;
    float qv[HD_];
    #pragma unroll
    for (int d = 0; d < HD_; d++) qv[d] = (float)qp[d];

    float m = -1e30f, l = 0.f;
    float acc[HD_];
    #pragma unroll
    for (int d = 0; d < HD_; d++) acc[d] = 0.f;

    dp_attn_accum(k1 + (long long)b * kBatch + h * HD_,
                  v1 + (long long)b * kBatch + h * HD_, kRow, kRow, n1, qv, m, l, acc);
    if (n2 > 0)
        dp_attn_accum(k2 + (long long)b * k2Batch + h * HD_,
                      v2 + (long long)b * k2Batch + h * HD_, k2Row, k2Row, n2, qv, m, l, acc);

    half_t* op = out + (long long)b * oBatch + (long long)t * oRow + h * HD_;
    float rl = 1.0f / l;
    #pragma unroll
    for (int d = 0; d < HD_; d++) op[d] = (half_t)(acc[d] * rl);
}

// ============================================================
// Cosine top-1 prompt selection: one block per batch element.
// ============================================================
__global__ void dp_select(const float* __restrict__ q, const float* __restrict__ ek,
                          int* __restrict__ idx) {
    const int b = blockIdx.x, tid = threadIdx.x;   // 128 threads
    __shared__ float sd[128], sk[128];
    const float* qb = q + (long long)b * D_;
    float qq = 0.f;
    for (int i = tid; i < D_; i += 128) qq += qb[i] * qb[i];
    sd[tid] = qq; __syncthreads();
    for (int off = 64; off; off >>= 1) { if (tid < off) sd[tid] += sd[tid + off]; __syncthreads(); }
    __shared__ float qnorm2;
    if (tid == 0) qnorm2 = sd[0];
    __syncthreads();
    __shared__ float best; __shared__ int bi;
    if (tid == 0) { best = -2.f; bi = 0; }
    __syncthreads();
    for (int e = 0; e < 10; e++) {
        float d = 0.f, kk = 0.f;
        for (int i = tid; i < D_; i += 128) {
            float kv = ek[(long long)e * D_ + i];
            d += qb[i] * kv; kk += kv * kv;
        }
        sd[tid] = d; sk[tid] = kk; __syncthreads();
        for (int off = 64; off; off >>= 1) {
            if (tid < off) { sd[tid] += sd[tid + off]; sk[tid] += sk[tid + off]; }
            __syncthreads();
        }
        if (tid == 0) {
            float cosv = sd[0] * rsqrtf(sk[0] * qnorm2 + 1e-12f);
            if (cosv > best) { best = cosv; bi = e; }
        }
        __syncthreads();
    }
    if (tid == 0) idx[b] = bi;
}

// ============================================================
// Build f16 prompt tokens: dst[b,t,d] = src[sel(b)][rowOff+t][d] + cls_pe[d]
// ============================================================
__global__ void dp_prompt_build(const float* __restrict__ src, long long idxStride,
                                const int* __restrict__ idx,
                                const float* __restrict__ cls_pe,
                                half_t* __restrict__ dst, int lenP, int rowOff) {
    long long i = (long long)blockIdx.x * blockDim.x + threadIdx.x;
    const long long total = (long long)B_ * lenP * D_;
    if (i >= total) return;
    int d = (int)(i % D_);
    long long r = i / D_;
    int t = (int)(r % lenP);
    int b = (int)(r / lenP);
    long long base = (idx ? (long long)idx[b] * idxStride : 0) + (long long)(rowOff + t) * D_;
    dst[i] = (half_t)(src[base + d] + cls_pe[d]);
}

// ============================================================
// head extraction: out[64,100] from padded [128-wide] scratch
// ============================================================
__global__ void dp_head_copy(const float* __restrict__ src, float* __restrict__ dst) {
    int i = blockIdx.x * blockDim.x + threadIdx.x;
    if (i >= B_ * NC_) return;
    int r = i / NC_, c = i % NC_;
    dst[i] = src[r * 128 + c];
}

// ============================================================
// host side
// ============================================================
static inline void dp_gemm(const half_t* A, int lda, const half_t* Bw, int ldb,
                           const float* bias, float* Cf, half_t* Ch, int ldc,
                           int M, int N, int K, int addC, int act, hipStream_t s) {
    dim3 g((N + 127) / 128, (M + 127) / 128), b(256);
    dp_gemm_wmma<<<g, b, 0, s>>>(A, lda, Bw, ldb, bias, Cf, Ch, ldc, K, addC, act);
}

static inline void dp_cvt(const float* s, half_t* d, long long n, hipStream_t st) {
    dp_cvt_f32_to_f16<<<(unsigned)((n + 255) / 256), 256, 0, st>>>(s, d, n);
}

extern "C" void kernel_launch(void* const* d_in, const int* in_sizes, int n_in,
                              void* d_out, int out_size, void* d_ws, size_t ws_size,
                              hipStream_t stream) {
    (void)in_sizes; (void)n_in; (void)out_size; (void)ws_size;
    const float* inputs    = (const float*)d_in[0];
    const float* patch_w   = (const float*)d_in[1];
    const float* patch_b   = (const float*)d_in[2];
    const float* cls_token = (const float*)d_in[3];
    const float* pos_embed = (const float*)d_in[4];
    const float* ln1_g = (const float*)d_in[5];
    const float* ln1_b = (const float*)d_in[6];
    const float* qkv_w = (const float*)d_in[7];
    const float* qkv_b = (const float*)d_in[8];
    const float* proj_w = (const float*)d_in[9];
    const float* proj_b = (const float*)d_in[10];
    const float* ln2_g = (const float*)d_in[11];
    const float* ln2_b = (const float*)d_in[12];
    const float* fc1_w = (const float*)d_in[13];
    const float* fc1_b = (const float*)d_in[14];
    const float* fc2_w = (const float*)d_in[15];
    const float* fc2_b = (const float*)d_in[16];
    const float* norm_g = (const float*)d_in[17];
    const float* norm_b = (const float*)d_in[18];
    const float* head_w = (const float*)d_in[19];
    const float* head_b = (const float*)d_in[20];
    const float* g_prompt = (const float*)d_in[21];
    const float* e_key    = (const float*)d_in[22];
    const float* e_prompt = (const float*)d_in[23];
    float* out = (float*)d_out;

    // ---- workspace carve-out (256B aligned) ----
    char* wsp = (char*)d_ws;
    auto alloc = [&](size_t bytes) -> void* {
        void* r = (void*)wsp;
        wsp += (bytes + 255) & ~(size_t)255;
        return r;
    };
    half_t* w_patch = (half_t*)alloc((size_t)D_ * 768 * 2);
    half_t* w_qkv   = (half_t*)alloc((size_t)L_ * 3 * D_ * D_ * 2);
    half_t* w_proj  = (half_t*)alloc((size_t)L_ * D_ * D_ * 2);
    half_t* w_fc1   = (half_t*)alloc((size_t)L_ * FF_ * D_ * 2);
    half_t* w_fc2   = (half_t*)alloc((size_t)L_ * D_ * FF_ * 2);
    half_t* w_head  = (half_t*)alloc((size_t)128 * D_ * 2);       // zero-padded to N=128
    float*  hb_pad  = (float*) alloc((size_t)128 * 4);            // zero-padded head bias
    half_t* im2c    = (half_t*)alloc((size_t)MP_ * 768 * 2);
    half_t* patches = (half_t*)alloc((size_t)MP_ * D_ * 2);
    float*  tok_pe  = (float*) alloc((size_t)M1_ * D_ * 4);
    float*  x       = (float*) alloc((size_t)MPAD_ * D_ * 4);
    half_t* xn      = (half_t*)alloc((size_t)MPAD_ * D_ * 2);
    half_t* qkvb    = (half_t*)alloc((size_t)MPAD_ * 3 * D_ * 2);
    half_t* attno   = (half_t*)alloc((size_t)MPAD_ * D_ * 2);
    half_t* hbuf    = (half_t*)alloc((size_t)MPAD_ * FF_ * 2);
    float*  xq      = (float*) alloc((size_t)B_ * D_ * 4);
    int*    idx     = (int*)   alloc((size_t)B_ * 4);
    half_t* cls16   = (half_t*)alloc((size_t)128 * D_ * 2);       // padded to 128 rows
    float*  out_pad = (float*) alloc((size_t)128 * 128 * 4);
    half_t* pk16    = (half_t*)alloc((size_t)B_ * 20 * D_ * 2);   // >= 384 padded rows
    half_t* pv16    = (half_t*)alloc((size_t)B_ * 20 * D_ * 2);
    half_t* kp      = (half_t*)alloc((size_t)B_ * 20 * D_ * 2);
    half_t* vp      = (half_t*)alloc((size_t)B_ * 20 * D_ * 2);

    // ---- weights -> f16 (head zero-padded to 128 rows) ----
    dp_cvt(patch_w, w_patch, (long long)D_ * 768, stream);
    dp_cvt(qkv_w,   w_qkv,   (long long)L_ * 3 * D_ * D_, stream);
    dp_cvt(proj_w,  w_proj,  (long long)L_ * D_ * D_, stream);
    dp_cvt(fc1_w,   w_fc1,   (long long)L_ * FF_ * D_, stream);
    dp_cvt(fc2_w,   w_fc2,   (long long)L_ * D_ * FF_, stream);
    {
        long long ntot = (long long)128 * D_;
        dp_cvt_pad<<<(unsigned)((ntot + 255) / 256), 256, 0, stream>>>(head_w, w_head,
                                                                       (long long)NC_ * D_, ntot);
        dp_pad_f32<<<1, 128, 0, stream>>>(head_b, hb_pad, NC_, 128);
    }

    // ---- patch embed ----
    {
        long long n = (long long)MP_ * 768;
        dp_im2col<<<(unsigned)((n + 255) / 256), 256, 0, stream>>>(inputs, im2c);
    }
    dp_gemm(im2c, 768, w_patch, 768, patch_b, nullptr, patches, D_, MP_, D_, 768, 0, 0, stream);
    {
        long long n = (long long)M1_ * D_;
        dp_build_tokens<<<(unsigned)((n + 255) / 256), 256, 0, stream>>>(patches, cls_token, pos_embed, tok_pe);
    }
    (void)hipMemcpyAsync(x, tok_pe, (size_t)M1_ * D_ * 4, hipMemcpyDeviceToDevice, stream);

    const long long qkvBatch = (long long)NT_ * 3 * D_;
    const long long xBatch   = (long long)NT_ * D_;
    const int attn_total = B_ * H_ * NT_;
    const int attn_blocks = (attn_total + 255) / 256;

    for (int pass = 0; pass < 2; pass++) {
        for (int i = 0; i < L_; i++) {
            // LN1 -> xn (f16)
            dp_layernorm<<<M1_, 256, 0, stream>>>(x, D_, ln1_g + i * D_, ln1_b + i * D_,
                                                  xn, D_, nullptr, 0);
            // fused QKV
            dp_gemm(xn, D_, w_qkv + (size_t)i * 3 * D_ * D_, D_, qkv_b + (size_t)i * 3 * D_,
                    nullptr, qkvb, 3 * D_, M1_, 3 * D_, D_, 0, 0, stream);

            // prefix prompts (pass 2 only)
            int lenP = 0;
            if (pass == 1) {
                const float* psrc = nullptr; long long pstride = 0; const int* pidx = nullptr;
                int offK = 0, offV = 0;
                if (i == 0 || i == 1) {            // g-prompt layers
                    lenP = 5; int j = i;
                    psrc = g_prompt; pstride = 0; pidx = nullptr;
                    offK = 2 * j * 5; offV = (2 * j + 1) * 5;
                } else if (i >= 2 && i <= 4) {     // e-prompt layers
                    lenP = 20; int j = i - 2;
                    psrc = e_prompt; pstride = (long long)120 * D_; pidx = idx;
                    offK = 2 * j * 20; offV = (2 * j + 1) * 20;
                }
                if (lenP) {
                    long long n = (long long)B_ * lenP * D_;
                    unsigned blks = (unsigned)((n + 255) / 256);
                    dp_prompt_build<<<blks, 256, 0, stream>>>(psrc, pstride, pidx, pos_embed, pk16, lenP, offK);
                    dp_prompt_build<<<blks, 256, 0, stream>>>(psrc, pstride, pidx, pos_embed, pv16, lenP, offV);
                    dp_gemm(pk16, D_, w_qkv + (size_t)i * 3 * D_ * D_ + (size_t)D_ * D_, D_,
                            qkv_b + (size_t)i * 3 * D_ + D_, nullptr, kp, D_,
                            B_ * lenP, D_, D_, 0, 0, stream);
                    dp_gemm(pv16, D_, w_qkv + (size_t)i * 3 * D_ * D_ + (size_t)2 * D_ * D_, D_,
                            qkv_b + (size_t)i * 3 * D_ + 2 * D_, nullptr, vp, D_,
                            B_ * lenP, D_, D_, 0, 0, stream);
                }
            }

            // attention
            dp_attention<<<attn_blocks, 256, 0, stream>>>(
                qkvb, qkvb + D_, qkvb + 2 * D_,
                lenP ? kp : nullptr, lenP ? vp : nullptr, attno,
                NT_, lenP,
                3 * D_, qkvBatch,
                3 * D_, qkvBatch,
                D_, (long long)lenP * D_,
                D_, xBatch);

            // proj (+residual into x)
            dp_gemm(attno, D_, w_proj + (size_t)i * D_ * D_, D_, proj_b + (size_t)i * D_,
                    x, nullptr, D_, M1_, D_, D_, 1, 0, stream);
            // LN2 -> xn
            dp_layernorm<<<M1_, 256, 0, stream>>>(x, D_, ln2_g + i * D_, ln2_b + i * D_,
                                                  xn, D_, nullptr, 0);
            // fc1 + exact GELU -> hbuf (f16)
            dp_gemm(xn, D_, w_fc1 + (size_t)i * FF_ * D_, D_, fc1_b + (size_t)i * FF_,
                    nullptr, hbuf, FF_, M1_, FF_, D_, 0, 1, stream);
            // fc2 (+residual into x)
            dp_gemm(hbuf, FF_, w_fc2 + (size_t)i * D_ * FF_, FF_, fc2_b + (size_t)i * D_,
                    x, nullptr, D_, M1_, D_, FF_, 1, 0, stream);
        }
        if (pass == 0) {
            dp_layernorm<<<B_, 256, 0, stream>>>(x, (long long)NT_ * D_, norm_g, norm_b,
                                                 nullptr, 0, xq, D_);
            dp_select<<<B_, 128, 0, stream>>>(xq, e_key, idx);
            (void)hipMemcpyAsync(x, tok_pe, (size_t)M1_ * D_ * 4, hipMemcpyDeviceToDevice, stream);
        }
    }

    // final LN on cls rows -> f16, then head (padded, unguarded) + extract
    dp_layernorm<<<B_, 256, 0, stream>>>(x, (long long)NT_ * D_, norm_g, norm_b,
                                         cls16, D_, nullptr, 0);
    dp_gemm(cls16, D_, w_head, D_, hb_pad, out_pad, nullptr, 128, B_, 128, D_, 0, 0, stream);
    dp_head_copy<<<(B_ * NC_ + 255) / 256, 256, 0, stream>>>(out_pad, out);
}
